// CasualEncoder_45767171506197
// MI455X (gfx1250) — matrix operand order repaired and checked by reading, not compiled, not run on hardware
//
#include <hip/hip_runtime.h>

// ---------------------------------------------------------------------------
// Neural CDE forward for MI455X (gfx1250, wave32, WMMA bf16 16x16x32).
// B=8192, L=64, C=8, H=256, N_STEPS=32. All RK4 sample times fall inside
// spline segment 0, so only segment-0 derivative coefficients are needed.
// Round 2: A-fragments cached in registers across N-tiles (kills redundant
// ds_load traffic seen in round-1 asm), fast tanh/softplus, vectorized
// bf16 stage-prep stores.
// ---------------------------------------------------------------------------

#define B_SZ   8192
#define L_SZ   64
#define C_SZ   8
#define H_SZ   256
#define NSTEPS 32
#define EPSLN  1e-5f

typedef __attribute__((ext_vector_type(16))) __bf16 v16bf;
typedef __attribute__((ext_vector_type(8)))  __bf16 v8bf;
typedef __attribute__((ext_vector_type(8)))  float  v8f;
typedef __attribute__((ext_vector_type(4)))  float  v4f;

union FragU { v16bf v; v8bf h[2]; };
union Pack8 { v8bf v; __bf16 e[8]; };

struct AFrags { FragU a[8]; };   // full K=256 row fragment set for one wave

// Fast device math (results feed bf16 GEMMs / f32 output; rcp+exp HW ops).
__device__ __forceinline__ float fast_tanh(float x) {
  float e = __expf(2.0f * x);
  return 1.0f - 2.0f * __builtin_amdgcn_rcpf(e + 1.0f);
}
__device__ __forceinline__ float fast_softplus(float v) {
  return (v > 20.f) ? v : __logf(1.0f + __expf(v));
}

// Load the 8 A-fragments (16x256 bf16 act tile) for this lane once.
// Layout (16-bit A 16x32, matching packed B): m = lane&15, half = lane>>4,
// K runs [kt*32+16*half, +16) contiguously.
__device__ __forceinline__ void load_afrags(const __bf16* __restrict__ act,
                                            int lane, AFrags& A) {
  const int m = lane & 15, half = lane >> 4;
#pragma unroll
  for (int kt = 0; kt < 8; ++kt) {
    const __bf16* ap = act + m * 256 + kt * 32 + 16 * half;
    A.a[kt].h[0] = *(const v8bf*)ap;
    A.a[kt].h[1] = *(const v8bf*)(ap + 8);
  }
}

// One 16x16 f32 tile: cached A-fragments x packed B (global, L2-resident).
// B packed so each lane reads 16 contiguous bf16 (32B) per k-tile.
__device__ __forceinline__ v8f wmma_tile(const AFrags& A,
                                         const __bf16* __restrict__ wp,
                                         int nt, int lane) {
  v8f acc = {};
#pragma unroll
  for (int kt = 0; kt < 8; ++kt) {
    FragU b;
    const __bf16* bp = wp + (((size_t)nt * 8 + kt) * 32 + lane) * 16;
    b.h[0] = *(const v8bf*)bp;
    b.h[1] = *(const v8bf*)(bp + 8);
    acc = __builtin_amdgcn_wmma_f32_16x16x32_bf16(false, A.a[kt].v, false, b.v,
                                                  (short)0, acc, false, false);
  }
  return acc;
}

// LayerNorm (+bias) + softplus over 256 features/row: 16 lanes per row,
// 16 elements per lane, shuffle-reduce within the 16-lane half-wave,
// vectorized bf16 write-back.
__device__ __forceinline__ void ln_softplus(const float* __restrict__ gout,
                                            __bf16* __restrict__ act,
                                            const float* __restrict__ bias,
                                            const float* __restrict__ g,
                                            const float* __restrict__ beta,
                                            int tid) {
  const int m = tid >> 4, seg = tid & 15;
  float vals[16];
  float s = 0.f, ss = 0.f;
#pragma unroll
  for (int j = 0; j < 16; ++j) {
    int n = seg * 16 + j;
    float v = gout[m * 256 + n] + bias[n];
    vals[j] = v; s += v; ss += v * v;
  }
#pragma unroll
  for (int o = 1; o < 16; o <<= 1) {
    s  += __shfl_xor(s, o);
    ss += __shfl_xor(ss, o);
  }
  float mu   = s * (1.0f / 256.0f);
  float var  = ss * (1.0f / 256.0f) - mu * mu;
  float rstd = rsqrtf(var + EPSLN);
#pragma unroll
  for (int h = 0; h < 2; ++h) {
    Pack8 u;
#pragma unroll
    for (int j = 0; j < 8; ++j) {
      int n = seg * 16 + h * 8 + j;
      float v = (vals[h * 8 + j] - mu) * rstd * g[n] + beta[n];
      u.e[j] = (__bf16)fast_softplus(v);
    }
    *(v8bf*)&act[m * 256 + seg * 16 + h * 8] = u.v;
  }
}

// Dense 256->256 bf16 WMMA layer + LayerNorm + softplus (in: act, out: act).
__device__ __forceinline__ void dense_ln(__bf16* __restrict__ act,
                                         float* __restrict__ gout,
                                         const __bf16* __restrict__ wp,
                                         const float* __restrict__ bias,
                                         const float* __restrict__ g,
                                         const float* __restrict__ beta,
                                         int tid, int wid, int lane) {
  AFrags A;
  load_afrags(act, lane, A);
#pragma unroll
  for (int q = 0; q < 2; ++q) {
    int nt = wid * 2 + q;
    v8f acc = wmma_tile(A, wp, nt, lane);
    int n  = nt * 16 + (lane & 15);
    int mb = 8 * (lane >> 4);
#pragma unroll
    for (int i = 0; i < 8; ++i) gout[(mb + i) * 256 + n] = acc[i];
  }
  __syncthreads();
  ln_softplus(gout, act, bias, g, beta, tid);
  __syncthreads();
}

// Full vector field: act = bf16(z-ish) in, k (16x256 f32) out in gout.
// Layer3 (256->2048) fused with bias+tanh and the einsum over C=8 via
// 8-lane shuffle reduction; never materializes the 2048-wide tensor.
// A-fragments loaded once and reused across all 16 N-tiles of the wave.
__device__ __forceinline__ void vf_eval(__bf16* __restrict__ act,
                                        float* __restrict__ gout,
                                        const float* __restrict__ dxs,
                                        const __bf16* f_w0p, const float* f_b0,
                                        const float* f_g0,  const float* f_be0,
                                        const __bf16* f_w1p, const float* f_b1,
                                        const float* f_g1,  const float* f_be1,
                                        const __bf16* f_w2p, const float* f_b2,
                                        int tid, int wid, int lane) {
  dense_ln(act, gout, f_w0p, f_b0, f_g0, f_be0, tid, wid, lane);
  dense_ln(act, gout, f_w1p, f_b1, f_g1, f_be1, tid, wid, lane);
  AFrags A;
  load_afrags(act, lane, A);
  const int mb = 8 * (lane >> 4);
  const int c  = lane & 7;
  float dxl[8];
#pragma unroll
  for (int i = 0; i < 8; ++i) dxl[i] = dxs[(mb + i) * 8 + c];
#pragma unroll 2
  for (int q = 0; q < 16; ++q) {
    int nt = wid * 16 + q;                       // 0..127 across waves
    v8f acc = wmma_tile(A, f_w2p, nt, lane);
    int n = nt * 16 + (lane & 15);               // 0..2047
    int h = n >> 3;
    float bias = f_b2[n];
#pragma unroll
    for (int i = 0; i < 8; ++i) {
      float v = fast_tanh(acc[i] + bias) * dxl[i];
      v += __shfl_xor(v, 1);
      v += __shfl_xor(v, 2);
      v += __shfl_xor(v, 4);
      if ((lane & 7) == 0) gout[(mb + i) * 256 + h] = v;   // k[m][h]
    }
  }
  __syncthreads();
}

// ---------------------------------------------------------------------------
// Kernel 1: pack f32 weight W[K][N] (row-major, x @ W) into WMMA-B bf16
// fragment order: [nt][kt][lane][16], lane = 16*half + n%16,
// K = kt*32 + 16*half + j (lanes 0-15 hold K 0..15, lanes 16-31 K 16..31).
// ---------------------------------------------------------------------------
__global__ void pack_kernel(const float* __restrict__ W, __bf16* __restrict__ P,
                            int K, int N) {
  int e = blockIdx.x * 256 + threadIdx.x;
  int total = K * N;
  if (e >= total) return;
  int KT = K >> 5;
  int j    = e & 15;
  int l    = (e >> 4) & 31;
  int kt   = (e >> 9) % KT;
  int nt   = e / (512 * KT);
  int half = l >> 4;
  int n = nt * 16 + (l & 15);
  int k = kt * 32 + 16 * half + j;
  P[e] = (__bf16)W[(size_t)k * N + n];
}

// ---------------------------------------------------------------------------
// Kernel 2: per-(b,c) natural-cubic-spline Thomas solve (L=64, uniform grid);
// emits segment-0 derivative coefficients (b0, 2c0, 3d0).
// ---------------------------------------------------------------------------
__global__ void spline_kernel(const float* __restrict__ x,
                              const float* __restrict__ t,
                              float* __restrict__ coef) {
  int idx = blockIdx.x * blockDim.x + threadIdx.x;
  if (idx >= B_SZ * C_SZ) return;
  int b = idx >> 3, c = idx & 7;
  const float* p = x + (size_t)b * L_SZ * C_SZ + c;
  float path[L_SZ];
#pragma unroll
  for (int l = 0; l < L_SZ; ++l) path[l] = p[l * C_SZ];

  const float r  = 1.0f / (t[1] - t[0]);
  const float r2 = r * r;
  float dd[L_SZ], bb[L_SZ];
  dd[0] = 2.f * r;
  bb[0] = 3.f * (path[1] - path[0]) * r2;
#pragma unroll
  for (int i = 1; i < L_SZ; ++i) {
    float di   = (i == L_SZ - 1) ? 2.f * r : 4.f * r;
    float pdsm = 3.f * (path[i] - path[i - 1]) * r2;
    float pdsi = (i == L_SZ - 1) ? 0.f : 3.f * (path[i + 1] - path[i]) * r2;
    float w = r / dd[i - 1];
    dd[i] = di - w * r;
    bb[i] = (pdsm + pdsi) - w * bb[i - 1];
  }
  float knext = bb[L_SZ - 1] / dd[L_SZ - 1];
  float k0v = 0.f, k1v = 0.f;
#pragma unroll
  for (int i = L_SZ - 2; i >= 0; --i) {
    float ki = (bb[i] - r * knext) / dd[i];
    if (i == 1) k1v = ki;
    if (i == 0) k0v = ki;
    knext = ki;
  }
  float spd0 = 6.f * (path[1] - path[0]);
  size_t o = (size_t)idx * 3;
  coef[o + 0] = k0v;                                        // b
  coef[o + 1] = (spd0 * r - 4.f * k0v - 2.f * k1v) * r;     // 2c
  coef[o + 2] = (-spd0 * r + 3.f * (k0v + k1v)) * r2;       // 3d
}

// ---------------------------------------------------------------------------
// Kernel 3: persistent forward. 512 blocks x 256 threads (8 wave32),
// each block owns 16 batch rows; z/kacc/act live in LDS for all 32 RK4 steps.
// ---------------------------------------------------------------------------
__global__ __launch_bounds__(256) void ncde_main(
    const float* __restrict__ t, const float* __restrict__ x,
    const float* __restrict__ coef,
    const float* e_w0, const float* e_b0, const float* e_g0, const float* e_be0,
    const __bf16* e_w1p, const float* e_b1, const float* e_g1, const float* e_be1,
    const __bf16* e_w2p, const float* e_b2,
    const __bf16* f_w0p, const float* f_b0, const float* f_g0, const float* f_be0,
    const __bf16* f_w1p, const float* f_b1, const float* f_g1, const float* f_be1,
    const __bf16* f_w2p, const float* f_b2,
    const __bf16* r_w0p, const float* r_b0, const float* r_g0, const float* r_be0,
    const __bf16* r_w1p, const float* r_b1, const float* r_g1, const float* r_be1,
    const __bf16* r_w2p, const float* r_b2,
    float* __restrict__ out) {
  __shared__ __align__(16) __bf16 act[16 * 256];   //  8 KB bf16 GEMM input
  __shared__ __align__(16) float  gout[16 * 256];  // 16 KB GEMM out / k
  __shared__ __align__(16) float  zbuf[16 * 256];  // 16 KB state z
  __shared__ __align__(16) float  kacc[16 * 256];  // 16 KB RK4 accumulator
  __shared__ float x0[16 * 8];
  __shared__ float spl[16 * 8 * 3];
  __shared__ float dxs[16 * 8];

  const int tid  = threadIdx.x;
  const int wid  = tid >> 5;
  const int lane = tid & 31;
  const int row0 = blockIdx.x * 16;
  const float dtv = (t[1] - t[0]) * (1.0f / (float)NSTEPS);

  if (tid < 128) {
    int m = tid >> 3, c = tid & 7;
    x0[m * 8 + c] = x[(size_t)(row0 + m) * L_SZ * C_SZ + c];
    size_t ci = ((size_t)(row0 + m) * C_SZ + c) * 3;
    spl[(m * 8 + c) * 3 + 0] = coef[ci + 0];
    spl[(m * 8 + c) * 3 + 1] = coef[ci + 1];
    spl[(m * 8 + c) * 3 + 2] = coef[ci + 2];
  }
  __syncthreads();

  const int lm = tid >> 4, seg = tid & 15;

  // ---- embed layer0 (8->256): tiny K, plain VALU ----
#pragma unroll
  for (int j = 0; j < 16; ++j) {
    int n = seg * 16 + j;
    float acc = 0.f;
#pragma unroll
    for (int c = 0; c < 8; ++c) acc += x0[lm * 8 + c] * e_w0[c * 256 + n];
    gout[lm * 256 + n] = acc;
  }
  __syncthreads();
  ln_softplus(gout, act, e_b0, e_g0, e_be0, tid);
  __syncthreads();
  dense_ln(act, gout, e_w1p, e_b1, e_g1, e_be1, tid, wid, lane);
  // embed layer2 (plain) -> z
  {
    AFrags A;
    load_afrags(act, lane, A);
#pragma unroll
    for (int q = 0; q < 2; ++q) {
      int nt = wid * 2 + q;
      v8f acc = wmma_tile(A, e_w2p, nt, lane);
      int n = nt * 16 + (lane & 15);
      float bias = e_b2[n];
      int mb = 8 * (lane >> 4);
#pragma unroll
      for (int i = 0; i < 8; ++i) zbuf[(mb + i) * 256 + n] = acc[i] + bias;
    }
  }
  __syncthreads();

  // ---- RK4 over segment 0 ----
  const float dt6 = dtv * (1.0f / 6.0f);
  for (int s = 0; s < NSTEPS; ++s) {
    const float f1 = s * dtv, f2 = f1 + 0.5f * dtv, f4 = f1 + dtv;

    // stage 1: act = bf16(z)
#pragma unroll
    for (int h = 0; h < 2; ++h) {
      Pack8 u;
#pragma unroll
      for (int j = 0; j < 8; ++j)
        u.e[j] = (__bf16)zbuf[lm * 256 + seg * 16 + h * 8 + j];
      *(v8bf*)&act[lm * 256 + seg * 16 + h * 8] = u.v;
    }
    if (tid < 128) {
      int m = tid >> 3, c = tid & 7;
      const float* sp = &spl[(m * 8 + c) * 3];
      dxs[m * 8 + c] = sp[0] + (sp[1] + sp[2] * f1) * f1;
    }
    __syncthreads();
    vf_eval(act, gout, dxs, f_w0p, f_b0, f_g0, f_be0,
            f_w1p, f_b1, f_g1, f_be1, f_w2p, f_b2, tid, wid, lane);

    // stage 2: kacc = k1; act = bf16(z + dt/2*k1)
#pragma unroll
    for (int h = 0; h < 2; ++h) {
      Pack8 u;
#pragma unroll
      for (int j = 0; j < 8; ++j) {
        int n = seg * 16 + h * 8 + j;
        float kv = gout[lm * 256 + n];
        kacc[lm * 256 + n] = kv;
        u.e[j] = (__bf16)(zbuf[lm * 256 + n] + 0.5f * dtv * kv);
      }
      *(v8bf*)&act[lm * 256 + seg * 16 + h * 8] = u.v;
    }
    if (tid < 128) {
      int m = tid >> 3, c = tid & 7;
      const float* sp = &spl[(m * 8 + c) * 3];
      dxs[m * 8 + c] = sp[0] + (sp[1] + sp[2] * f2) * f2;
    }
    __syncthreads();
    vf_eval(act, gout, dxs, f_w0p, f_b0, f_g0, f_be0,
            f_w1p, f_b1, f_g1, f_be1, f_w2p, f_b2, tid, wid, lane);

    // stage 3: kacc += 2*k2; act = bf16(z + dt/2*k2)  (dx unchanged)
#pragma unroll
    for (int h = 0; h < 2; ++h) {
      Pack8 u;
#pragma unroll
      for (int j = 0; j < 8; ++j) {
        int n = seg * 16 + h * 8 + j;
        float kv = gout[lm * 256 + n];
        kacc[lm * 256 + n] += 2.f * kv;
        u.e[j] = (__bf16)(zbuf[lm * 256 + n] + 0.5f * dtv * kv);
      }
      *(v8bf*)&act[lm * 256 + seg * 16 + h * 8] = u.v;
    }
    __syncthreads();
    vf_eval(act, gout, dxs, f_w0p, f_b0, f_g0, f_be0,
            f_w1p, f_b1, f_g1, f_be1, f_w2p, f_b2, tid, wid, lane);

    // stage 4: kacc += 2*k3; act = bf16(z + dt*k3)
#pragma unroll
    for (int h = 0; h < 2; ++h) {
      Pack8 u;
#pragma unroll
      for (int j = 0; j < 8; ++j) {
        int n = seg * 16 + h * 8 + j;
        float kv = gout[lm * 256 + n];
        kacc[lm * 256 + n] += 2.f * kv;
        u.e[j] = (__bf16)(zbuf[lm * 256 + n] + dtv * kv);
      }
      *(v8bf*)&act[lm * 256 + seg * 16 + h * 8] = u.v;
    }
    if (tid < 128) {
      int m = tid >> 3, c = tid & 7;
      const float* sp = &spl[(m * 8 + c) * 3];
      dxs[m * 8 + c] = sp[0] + (sp[1] + sp[2] * f4) * f4;
    }
    __syncthreads();
    vf_eval(act, gout, dxs, f_w0p, f_b0, f_g0, f_be0,
            f_w1p, f_b1, f_g1, f_be1, f_w2p, f_b2, tid, wid, lane);

    // z += dt/6 * (k1 + 2k2 + 2k3 + k4)
#pragma unroll
    for (int j = 0; j < 16; ++j) {
      int n = seg * 16 + j;
      zbuf[lm * 256 + n] += dt6 * (kacc[lm * 256 + n] + gout[lm * 256 + n]);
    }
    __syncthreads();
  }

  // ---- readout ----
#pragma unroll
  for (int h = 0; h < 2; ++h) {
    Pack8 u;
#pragma unroll
    for (int j = 0; j < 8; ++j)
      u.e[j] = (__bf16)zbuf[lm * 256 + seg * 16 + h * 8 + j];
    *(v8bf*)&act[lm * 256 + seg * 16 + h * 8] = u.v;
  }
  __syncthreads();
  dense_ln(act, gout, r_w0p, r_b0, r_g0, r_be0, tid, wid, lane);
  dense_ln(act, gout, r_w1p, r_b1, r_g1, r_be1, tid, wid, lane);
  {
    AFrags A;
    load_afrags(act, lane, A);
#pragma unroll
    for (int q = 0; q < 2; ++q) {
      int nt = wid * 2 + q;
      v8f acc = wmma_tile(A, r_w2p, nt, lane);
      int n = nt * 16 + (lane & 15);
      float bias = r_b2[n];
      int mb = 8 * (lane >> 4);
#pragma unroll
      for (int i = 0; i < 8; ++i)
        out[(size_t)(row0 + mb + i) * 256 + n] = acc[i] + bias;
    }
  }
}

// ---------------------------------------------------------------------------
// Host-side launch. Input order assumes pytree flattening of the params dict
// (sorted keys): t, x, then per layer {b, beta, g, w} for embed[0..2],
// f[0..2], readout[0..2].
// ---------------------------------------------------------------------------
extern "C" void kernel_launch(void* const* d_in, const int* in_sizes, int n_in,
                              void* d_out, int out_size, void* d_ws, size_t ws_size,
                              hipStream_t stream) {
  (void)in_sizes; (void)n_in; (void)out_size; (void)ws_size;
  const float* t = (const float*)d_in[0];
  const float* x = (const float*)d_in[1];
  auto P = [&](int i) { return (const float*)d_in[i]; };
  // indices: embed L0 b=2 beta=3 g=4 w=5 | L1 6..9 | L2 10..13
  //          f     L0 14..17 | L1 18..21 | L2 22..25
  //          read  L0 26..29 | L1 30..33 | L2 34..37

  char* ws = (char*)d_ws;
  float*  coef  = (float*)ws;                              // 8192*8*3 f32
  __bf16* e_w1p = (__bf16*)(ws +  786432);
  __bf16* e_w2p = (__bf16*)(ws +  917504);
  __bf16* f_w0p = (__bf16*)(ws + 1048576);
  __bf16* f_w1p = (__bf16*)(ws + 1179648);
  __bf16* f_w2p = (__bf16*)(ws + 1310720);                 // 256x2048
  __bf16* r_w0p = (__bf16*)(ws + 2359296);
  __bf16* r_w1p = (__bf16*)(ws + 2490368);
  __bf16* r_w2p = (__bf16*)(ws + 2621440);

  pack_kernel<<<256, 256, 0, stream>>>(P(9),  e_w1p, 256, 256);
  pack_kernel<<<256, 256, 0, stream>>>(P(13), e_w2p, 256, 256);
  pack_kernel<<<256, 256, 0, stream>>>(P(17), f_w0p, 256, 256);
  pack_kernel<<<256, 256, 0, stream>>>(P(21), f_w1p, 256, 256);
  pack_kernel<<<2048, 256, 0, stream>>>(P(25), f_w2p, 256, 2048);
  pack_kernel<<<256, 256, 0, stream>>>(P(29), r_w0p, 256, 256);
  pack_kernel<<<256, 256, 0, stream>>>(P(33), r_w1p, 256, 256);
  pack_kernel<<<256, 256, 0, stream>>>(P(37), r_w2p, 256, 256);

  spline_kernel<<<(B_SZ * C_SZ + 255) / 256, 256, 0, stream>>>(x, t, coef);

  ncde_main<<<B_SZ / 16, 256, 0, stream>>>(
      t, x, coef,
      P(5),  P(2),  P(4),  P(3),
      e_w1p, P(6),  P(8),  P(7),
      e_w2p, P(10),
      f_w0p, P(14), P(16), P(15),
      f_w1p, P(18), P(20), P(19),
      f_w2p, P(22),
      r_w0p, P(26), P(28), P(27),
      r_w1p, P(30), P(32), P(31),
      r_w2p, P(34),
      (float*)d_out);
}
